// CMAtten_Align_64012192579880
// MI455X (gfx1250) — matrix operand order repaired and checked by reading, not compiled, not run on hardware
//
#include <hip/hip_runtime.h>
#include <hip/hip_bf16.h>

// Problem dims (fixed by the reference)
#define BATCH 8
#define T1    2048
#define T2    2048
#define DDIM  256

// Tiling
#define ROWS  128                           // s1 rows per block (8 waves x 16 rows)
#define SPLIT 4                             // T2 column splits (parallelism)
#define COLS_PER_SPLIT  (T2 / SPLIT)        // 512
#define TILES_PER_SPLIT (COLS_PER_SPLIT/16) // 32
#define LDB   264                           // padded LDS row stride (bf16 elems)

typedef __attribute__((ext_vector_type(16))) __bf16 v16bf;
typedef __attribute__((ext_vector_type(8)))  __bf16 v8bf;
typedef __attribute__((ext_vector_type(4)))  __bf16 v4bf;
typedef __attribute__((ext_vector_type(8)))  float  v8f;
typedef __attribute__((ext_vector_type(4)))  float  vf4;

static __device__ __forceinline__ v8f wmma_bf16(v16bf a, v16bf b, v8f c) {
  // D = A*B + C, 16x16x32 bf16 -> f32
  return __builtin_amdgcn_wmma_f32_16x16x32_bf16(
      /*neg_a=*/false, a, /*neg_b=*/false, b,
      /*c_mod=*/(short)0, c, /*reuse_a=*/false, /*reuse_b=*/false);
}

// B-matrix fragment (32x16, K-major per column): lane<16 holds K in {kc*32+[0,8), kc*32+[16,24)},
// lanes>=16 hold +8 on both ranges -> two 16B LDS loads.
static __device__ __forceinline__ v16bf ld_b_frag(const __bf16* row, int kc, int hi16) {
  const int kb = kc * 32 + (hi16 ? 8 : 0);
  v8bf a = *(const v8bf*)(row + kb);
  v8bf b = *(const v8bf*)(row + kb + 16);
  return __builtin_shufflevector(a, b, 0,1,2,3,4,5,6,7,8,9,10,11,12,13,14,15);
}

extern "C" __global__ __launch_bounds__(256)
void cm_argmax_gemm(const float* __restrict__ s1, const float* __restrict__ s2,
                    float* __restrict__ pval, int* __restrict__ pidx)
{
  extern __shared__ __align__(16) char smem_raw[];
  __bf16* Bhi = (__bf16*)smem_raw;          // [16][LDB]
  __bf16* Blo = Bhi + 16 * LDB;             // [16][LDB]

  const int t    = threadIdx.x;
  const int lane = t & 31;
  const int w    = t >> 5;                  // wave id 0..7
  const int hi16 = lane & 16;
  const int lrow = lane & 15;
  const int b    = blockIdx.y;
  const int r0   = blockIdx.x * ROWS;
  const int sp   = blockIdx.z;
  const int c0   = sp * COLS_PER_SPLIT;

  // ---- A fragments: load this wave's 16 s1 rows, split f32 -> bf16 hi+lo,
  //      register-resident for the whole column sweep.
  // ISA A layout (16-bit, 16x32): lane<16 -> K in {kc*32+[0,8), kc*32+[16,24)},
  //                               lane>=16 -> +8 on both ranges.
  v16bf ahi[8], alo[8];
  {
    const float* arow = s1 + ((size_t)b * T1 + r0 + w * 16 + lrow) * DDIM;
    #pragma unroll
    for (int kc = 0; kc < 8; ++kc) {
      const int kb = kc * 32 + (hi16 ? 8 : 0);
      float fv[16];
      *(vf4*)(fv + 0)  = *(const vf4*)(arow + kb);
      *(vf4*)(fv + 4)  = *(const vf4*)(arow + kb + 4);
      *(vf4*)(fv + 8)  = *(const vf4*)(arow + kb + 16);
      *(vf4*)(fv + 12) = *(const vf4*)(arow + kb + 20);
      v16bf h, l;
      #pragma unroll
      for (int i = 0; i < 16; ++i) {
        __bf16 hh = (__bf16)fv[i];
        h[i] = hh;
        l[i] = (__bf16)(fv[i] - (float)hh);
      }
      ahi[kc] = h; alo[kc] = l;
    }
  }

  float bval[8]; int bidx[8];
  #pragma unroll
  for (int e = 0; e < 8; ++e) { bval[e] = -3.402823466e38f; bidx[e] = 0; }

  const __bf16* browh = Bhi + lrow * LDB;
  const __bf16* browl = Blo + lrow * LDB;

  for (int jt = 0; jt < TILES_PER_SPLIT; ++jt) {
    __syncthreads();                        // prior tile fully consumed
    // ---- Cooperative stage of 16 s2 rows (columns of s^T) as bf16 hi/lo
    {
      const float* src = s2 + ((size_t)b * T2 + c0 + jt * 16) * DDIM;
      #pragma unroll
      for (int i = 0; i < (16 * DDIM / 4) / 256; ++i) {   // 4 iterations
        const int q   = t + 256 * i;
        const int row = q >> 6;                           // 64 vf4 per row
        const int col = (q & 63) * 4;
        vf4 v = *(const vf4*)(src + row * DDIM + col);
        v4bf h, l;
        #pragma unroll
        for (int j = 0; j < 4; ++j) {
          __bf16 hh = (__bf16)v[j];
          h[j] = hh;
          l[j] = (__bf16)(v[j] - (float)hh);
        }
        *(v4bf*)(Bhi + row * LDB + col) = h;
        *(v4bf*)(Blo + row * LDB + col) = l;
      }
    }
    __syncthreads();

    // ---- 16x16x256 tile: 8 K-chunks x 3 WMMA (split-precision bf16x3)
    v8f acc = {};
    #pragma unroll
    for (int kc = 0; kc < 8; ++kc) {
      v16bf bh = ld_b_frag(browh, kc, hi16);
      v16bf bl = ld_b_frag(browl, kc, hi16);
      acc = wmma_bf16(alo[kc], bh, acc);    // lo*hi
      acc = wmma_bf16(ahi[kc], bl, acc);    // hi*lo
      acc = wmma_bf16(ahi[kc], bh, acc);    // hi*hi (dominant term last)
    }

    // ---- Per-lane running argmax (C layout: lane = column, VGPR = row)
    const int col = c0 + jt * 16 + lrow;
    #pragma unroll
    for (int e = 0; e < 8; ++e) {
      if (acc[e] > bval[e]) { bval[e] = acc[e]; bidx[e] = col; }
    }
  }

  // ---- Cross-lane argmax over the 16 columns held in each 16-lane half.
  // Tie-break: smaller index (matches jnp.argmax first-occurrence).
  #pragma unroll
  for (int e = 0; e < 8; ++e) {
    float v = bval[e]; int ix = bidx[e];
    #pragma unroll
    for (int m = 1; m <= 8; m <<= 1) {
      float v2 = __shfl_xor(v, m, 32);
      int   i2 = __shfl_xor(ix, m, 32);
      if (v2 > v || (v2 == v && i2 < ix)) { v = v2; ix = i2; }
    }
    if (lrow == 0) {
      const int row = r0 + w * 16 + e + (hi16 ? 8 : 0);
      const size_t o = (size_t)sp * (BATCH * T1) + (size_t)b * T1 + row;
      pval[o] = v; pidx[o] = ix;
    }
  }
}

// One block per (b, i): merge the SPLIT partial argmaxes, gather the u_tile
// row, and stream the one-hot row with nontemporal stores.
extern "C" __global__ __launch_bounds__(256)
void cm_finalize(const float* __restrict__ s2, const float* __restrict__ pval,
                 const int* __restrict__ pidx, float* __restrict__ u_tile,
                 float* __restrict__ a_w)
{
  const size_t r = blockIdx.x;              // 0 .. B*T1-1
  const int t = threadIdx.x;
  const int b = (int)(r / T1);

  float best = -3.402823466e38f; int idx = 0;
  #pragma unroll
  for (int s = 0; s < SPLIT; ++s) {
    const float v = pval[(size_t)s * (BATCH * T1) + r];
    const int   i = pidx[(size_t)s * (BATCH * T1) + r];
    if (v > best || (v == best && i < idx) || s == 0) { best = v; idx = i; }
  }

  // u_tile[b,i,:] = s2[b, idx, :]   (D = 256 = one float per thread)
  const float* srow = s2 + ((size_t)b * T2 + idx) * DDIM;
  u_tile[r * DDIM + t] = srow[t];

  // one-hot row: 2048 floats = 512 vf4, 2 per thread, nontemporal
  float* orow = a_w + r * T2;
  #pragma unroll
  for (int k = 0; k < T2 / (256 * 4); ++k) {
    const int col = (t + 256 * k) * 4;
    vf4 v = {0.f, 0.f, 0.f, 0.f};
    if (idx >= col && idx < col + 4) v[idx - col] = 1.0f;
    __builtin_nontemporal_store(v, (vf4*)(orow + col));
  }
}

extern "C" void kernel_launch(void* const* d_in, const int* in_sizes, int n_in,
                              void* d_out, int out_size, void* d_ws, size_t ws_size,
                              hipStream_t stream) {
  (void)in_sizes; (void)n_in; (void)out_size; (void)ws_size;
  const float* s1 = (const float*)d_in[0];
  const float* s2 = (const float*)d_in[1];

  float* u_tile = (float*)d_out;                                  // [B,T1,D]
  float* a_w    = u_tile + (size_t)BATCH * T1 * DDIM;             // [B,T1,T2]

  float* pval = (float*)d_ws;                                     // [SPLIT][B*T1]
  int*   pidx = (int*)(pval + (size_t)SPLIT * BATCH * T1);        // [SPLIT][B*T1]

  const dim3 g1(T1 / ROWS, BATCH, SPLIT);                         // 16 x 8 x 4
  const size_t smem = (size_t)(2 * 16 * LDB) * sizeof(__bf16);    // ~16.9 KB
  cm_argmax_gemm<<<g1, 256, smem, stream>>>(s1, s2, pval, pidx);

  cm_finalize<<<dim3(BATCH * T1), 256, 0, stream>>>(s2, pval, pidx, u_tile, a_w);
}
